// TopKSAE_22565758173711
// MI455X (gfx1250) — matrix operand order repaired and checked by reading, not compiled, not run on hardware
//
#include <hip/hip_runtime.h>
#include <hip/hip_bf16.h>

// Problem constants (TopK SAE): B=32768, D=1024, L=8192, K=64
#define B_   32768
#define D_   1024
#define L_   8192
#define KTOP 64

// Encoder GEMM tiling: 128x128 output tile per 256-thread block, K chunked by 32
#define TM  128
#define TN  128
#define TKC 32

typedef __attribute__((ext_vector_type(16))) __bf16          v16bf;
typedef __attribute__((ext_vector_type(8)))  float           v8f;
typedef __attribute__((ext_vector_type(8)))  unsigned short  ushort8;
typedef __attribute__((ext_vector_type(4)))  unsigned int    uint4v;
typedef __attribute__((ext_vector_type(8)))  int             int8v;
typedef __attribute__((ext_vector_type(4)))  int             int4v;

union BF16Frag { v16bf bf; ushort8 h[2]; };

__device__ __forceinline__ unsigned short f2bf(float f) {
  // round-to-nearest-even f32 -> bf16 (inputs finite)
  unsigned int u = __builtin_bit_cast(unsigned int, f);
  u += 0x7FFFu + ((u >> 16) & 1u);
  return (unsigned short)(u >> 16);
}

// Monotonic f32-bits <-> u32 key (larger float => larger key), pure bit ops
__device__ __forceinline__ unsigned int bits2key(unsigned int u) {
  return (u >> 31) ? ~u : (u | 0x80000000u);
}
__device__ __forceinline__ float key2f(unsigned int k) {
  unsigned int b = (k & 0x80000000u) ? (k & 0x7FFFFFFFu) : ~k;
  return __builtin_bit_cast(float, b);
}

// ---------------------------------------------------------------------------
// Kernel 1: latents = (x - b_pre) @ W_enc + b_enc  (dense, written into the
// sparse_latents region of d_out; kernel 2 sparsifies it in place).
// 8 waves/block, each wave owns a 2x4 grid of 16x16 WMMA subtiles.
// Register double-buffering: global loads for chunk k+1 overlap the WMMAs.
// ---------------------------------------------------------------------------
__global__ __launch_bounds__(256) void sae_encode_wmma(
    const float* __restrict__ x, const float* __restrict__ b_pre,
    const float* __restrict__ W_enc, const float* __restrict__ b_enc,
    float* __restrict__ latents)
{
  __shared__ __align__(16) unsigned short As[TM * TKC]; // [row][k]  bf16 bits
  __shared__ __align__(16) unsigned short Bs[TN * TKC]; // [col][k]  (transposed)

  const int tid  = threadIdx.x;
  const int lane = tid & 31;
  const int wave = tid >> 5;       // 0..7
  const int wr   = wave & 3;       // rows 32*wr .. +31
  const int wc   = wave >> 2;      // cols 64*wc .. +63
  const int l15  = lane & 15;
  const int kh   = lane >> 4;      // 0/1 half-wave

  const int m0 = blockIdx.y * TM;
  const int n0 = blockIdx.x * TN;

  // staging-pattern indices (constant across chunks)
  const int arow[4] = { (tid + 0)   >> 3, (tid + 256) >> 3,
                        (tid + 512) >> 3, (tid + 768) >> 3 };
  const int ac4 = tid & 7;                 // same for all j since 256 % 8 == 0
  const int bkk[4] = { (tid + 0)   >> 5, (tid + 256) >> 5,
                       (tid + 512) >> 5, (tid + 768) >> 5 };
  const int bc4 = tid & 31;                // same for all j since 256 % 32 == 0

  v8f acc[2][4];
  const v8f vzero = {0.f, 0.f, 0.f, 0.f, 0.f, 0.f, 0.f, 0.f};
#pragma unroll
  for (int r = 0; r < 2; ++r)
#pragma unroll
    for (int c = 0; c < 4; ++c) acc[r][c] = vzero;

  float4 xa[4], wb[4];
  // prologue: load chunk 0 into registers (A already centered by b_pre)
#pragma unroll
  for (int j = 0; j < 4; ++j) {
    const float4 xv = *(const float4*)(x + (size_t)(m0 + arow[j]) * D_ + ac4 * 4);
    const float4 bp = *(const float4*)(b_pre + ac4 * 4);
    xa[j].x = xv.x - bp.x; xa[j].y = xv.y - bp.y;
    xa[j].z = xv.z - bp.z; xa[j].w = xv.w - bp.w;
    wb[j] = *(const float4*)(W_enc + (size_t)bkk[j] * L_ + n0 + bc4 * 4);
  }

  for (int k0 = 0; k0 < D_; k0 += TKC) {
    // ---- stage current chunk from registers into LDS (f32 -> bf16) ----
#pragma unroll
    for (int j = 0; j < 4; ++j) {
      int ab = arow[j] * TKC + ac4 * 4;
      As[ab + 0] = f2bf(xa[j].x); As[ab + 1] = f2bf(xa[j].y);
      As[ab + 2] = f2bf(xa[j].z); As[ab + 3] = f2bf(xa[j].w);
      Bs[(bc4 * 4 + 0) * TKC + bkk[j]] = f2bf(wb[j].x);
      Bs[(bc4 * 4 + 1) * TKC + bkk[j]] = f2bf(wb[j].y);
      Bs[(bc4 * 4 + 2) * TKC + bkk[j]] = f2bf(wb[j].z);
      Bs[(bc4 * 4 + 3) * TKC + bkk[j]] = f2bf(wb[j].w);
    }
    __syncthreads();

    // ---- issue global loads for next chunk (latency hidden by WMMAs) ----
    const int kn = k0 + TKC;
    if (kn < D_) {
#pragma unroll
      for (int j = 0; j < 4; ++j) {
        const float4 xv = *(const float4*)(x + (size_t)(m0 + arow[j]) * D_ + kn + ac4 * 4);
        const float4 bp = *(const float4*)(b_pre + kn + ac4 * 4);
        xa[j].x = xv.x - bp.x; xa[j].y = xv.y - bp.y;
        xa[j].z = xv.z - bp.z; xa[j].w = xv.w - bp.w;
        wb[j] = *(const float4*)(W_enc + (size_t)(kn + bkk[j]) * L_ + n0 + bc4 * 4);
      }
    }

    // ---- fragments per ISA VGPR layouts ----
    // A 16x32: lanes 0-15 (row=l15) K {0..7},{16..23}; lanes 16-31 K {8..15},{24..31}
    BF16Frag af[2], bfr[4];
#pragma unroll
    for (int r = 0; r < 2; ++r) {
      int rrow = 32 * wr + 16 * r + l15;
      af[r].h[0] = *(const ushort8*)&As[rrow * TKC + kh * 8];
      af[r].h[1] = *(const ushort8*)&As[rrow * TKC + 16 + kh * 8];
    }
  // B 32x16: lanes 0-15 (col=l15) K {0..7},{16..23}; lanes 16-31 K {8..15},{24..31}
#pragma unroll
    for (int c = 0; c < 4; ++c) {
      int col = 64 * wc + 16 * c + l15;
      bfr[c].h[0] = *(const ushort8*)&Bs[col * TKC + kh * 8];
      bfr[c].h[1] = *(const ushort8*)&Bs[col * TKC + 16 + kh * 8];
    }
#pragma unroll
    for (int r = 0; r < 2; ++r)
#pragma unroll
      for (int c = 0; c < 4; ++c)
        acc[r][c] = __builtin_amdgcn_wmma_f32_16x16x32_bf16(
            false, af[r].bf, false, bfr[c].bf, (short)0, acc[r][c], false, false);
    __syncthreads();
  }

  // ---- epilogue: + b_enc, store f32.  C/D layout: VGPR v -> M=v+8*kh, N=l15
#pragma unroll
  for (int c = 0; c < 4; ++c) {
    int col = n0 + 64 * wc + 16 * c + l15;
    float be = b_enc[col];
#pragma unroll
    for (int r = 0; r < 2; ++r) {
      int rowBase = m0 + 32 * wr + 16 * r + 8 * kh;
#pragma unroll
      for (int v = 0; v < 8; ++v)
        latents[(size_t)(rowBase + v) * L_ + col] = acc[r][c][v] + be;
    }
  }
}

// ---------------------------------------------------------------------------
// Kernel 2: per-row exact top-64 + in-place sparsification + sparse decode.
// One 256-thread block per row; thread t owns strided elements i = t + 256*j
// (LDS bank-conflict-free, row stays contiguous for TDM).
// Row in/out moved by the Tensor Data Mover (tensor_load_to_lds /
// tensor_store_from_lds); the TDM store overlaps the decode GEMV.
// All ranks (ties, slots) come from prefix scans => bitwise-deterministic.
// ---------------------------------------------------------------------------
__global__ __launch_bounds__(256) void sae_topk_decode(
    const float* __restrict__ b_pre, const float* __restrict__ W_dec,
    const float* __restrict__ b_dec, float* __restrict__ recon,
    float* __restrict__ latents)
{
  __shared__ __align__(16) unsigned int kb[L_];  // 32 KB: floats -> keys -> floats
  __shared__ int   red[256];
  __shared__ float selV[KTOP];
  __shared__ int   selI[KTOP];

  const int t   = threadIdx.x;
  const int row = blockIdx.x;
  float* lat = latents + (size_t)row * L_;
  const unsigned ldsAddr = (unsigned)(size_t)(void*)kb;   // low 32 bits = LDS offset
  const unsigned long long ga = (unsigned long long)(size_t)lat;

  // ---- load row into LDS via the Tensor Data Mover ----
#if __has_builtin(__builtin_amdgcn_tensor_load_to_lds)
  if (t < 32) {
    // D# group0: count=1 | lds_addr | global_addr[56:0] | type=2
    uint4v g0 = { 1u, ldsAddr, (unsigned)ga,
                  ((unsigned)(ga >> 32) & 0x01FFFFFFu) | 0x80000000u };
    // D# group1: data_size=4B; tensor_dim0=8192, tensor_dim1=1;
    //            tile_dim0=8192, tile_dim1=1; dim0_stride=8192
    int8v g1 = { 0x00020000, (int)(8192u << 16), 0x00010000,
                 (int)(8192u << 16), 1, 8192, 0, 0 };
    int4v gz4 = { 0, 0, 0, 0 };
    int8v gz8 = { 0, 0, 0, 0, 0, 0, 0, 0 };
    __builtin_amdgcn_tensor_load_to_lds(g0, g1, gz4, gz4, gz8, 0);
    __builtin_amdgcn_s_wait_tensorcnt(0);
  }
#else
#pragma unroll
  for (int j = 0; j < 8; ++j) {
    int i4 = t + 256 * j;                       // 2048 float4 = whole row
    const float4 v = *(const float4*)(lat + i4 * 4);
    kb[i4 * 4 + 0] = __builtin_bit_cast(unsigned int, v.x);
    kb[i4 * 4 + 1] = __builtin_bit_cast(unsigned int, v.y);
    kb[i4 * 4 + 2] = __builtin_bit_cast(unsigned int, v.z);
    kb[i4 * 4 + 3] = __builtin_bit_cast(unsigned int, v.w);
  }
#endif
  __syncthreads();

  // ---- convert to sortable keys in place (pure bit ops) ----
#pragma unroll
  for (int j = 0; j < 32; ++j) {
    int i = t + 256 * j;
    kb[i] = bits2key(kb[i]);
  }
  __syncthreads();

  // ---- bitwise binary search for T = 64th-largest key ----
  unsigned int prefix = 0;
  for (int bit = 31; bit >= 0; --bit) {
    unsigned int cand = prefix | (1u << bit);
    int c = 0;
#pragma unroll
    for (int j = 0; j < 32; ++j) c += (kb[t + 256 * j] >= cand) ? 1 : 0;
    red[t] = c;
    __syncthreads();
    for (int s = 128; s > 0; s >>= 1) {
      if (t < s) red[t] += red[t + s];
      __syncthreads();
    }
    if (red[0] >= KTOP) prefix = cand;
    __syncthreads();
  }
  const unsigned int T = prefix;

  // ---- per-thread strictly-greater / tie counts ----
  int cgt = 0, ctie = 0;
#pragma unroll
  for (int j = 0; j < 32; ++j) {
    unsigned int k = kb[t + 256 * j];
    cgt  += (k > T) ? 1 : 0;
    ctie += (k == T) ? 1 : 0;
  }
  red[t] = cgt;
  __syncthreads();
  for (int s = 128; s > 0; s >>= 1) {
    if (t < s) red[t] += red[t + s];
    __syncthreads();
  }
  const int numTies = KTOP - red[0];
  __syncthreads();

  // inclusive scan of tie counts -> deterministic tie base per thread
  red[t] = ctie;
  __syncthreads();
  for (int s = 1; s < 256; s <<= 1) {
    int v = (t >= s) ? red[t - s] : 0;
    __syncthreads();
    red[t] += v;
    __syncthreads();
  }
  const int tieBase = red[t] - ctie;
  const int myTiesUsed = min(max(numTies - tieBase, 0), ctie);
  __syncthreads();

  // inclusive scan of selected counts -> deterministic slot base per thread
  const int selCnt = cgt + myTiesUsed;
  red[t] = selCnt;
  __syncthreads();
  for (int s = 1; s < 256; s <<= 1) {
    int v = (t >= s) ? red[t - s] : 0;
    __syncthreads();
    red[t] += v;
    __syncthreads();
  }
  const int selBase  = red[t] - selCnt;
  const int totalSel = red[255];

  // ---- select, collect (val,idx) into fixed slots, rewrite row in LDS ----
  int tieRank = tieBase;
  int slot    = selBase;
#pragma unroll
  for (int j = 0; j < 32; ++j) {
    int i = t + 256 * j;
    unsigned int k = kb[i];
    bool sel = (k > T);
    if (k == T) {
      if (tieRank < numTies) sel = true;
      ++tieRank;
    }
    float fv = sel ? key2f(k) : 0.f;
    kb[i] = __builtin_bit_cast(unsigned int, fv);
    if (sel && slot < KTOP) { selV[slot] = fv; selI[slot] = i; ++slot; }
  }
  __syncthreads();

  // ---- DMA sparsified row back to global; overlaps the decode below ----
#if __has_builtin(__builtin_amdgcn_tensor_store_from_lds)
  if (t < 32) {
    uint4v g0 = { 1u, ldsAddr, (unsigned)ga,
                  ((unsigned)(ga >> 32) & 0x01FFFFFFu) | 0x80000000u };
    int8v g1 = { 0x00020000, (int)(8192u << 16), 0x00010000,
                 (int)(8192u << 16), 1, 8192, 0, 0 };
    int4v gz4 = { 0, 0, 0, 0 };
    int8v gz8 = { 0, 0, 0, 0, 0, 0, 0, 0 };
    __builtin_amdgcn_tensor_store_from_lds(g0, g1, gz4, gz4, gz8, 0);
    // no wait needed: decode doesn't touch kb; S_ENDPGM waits for TENSORcnt
  }
#else
#pragma unroll
  for (int j = 0; j < 32; ++j) {
    int i = t + 256 * j;
    lat[i] = __builtin_bit_cast(float, kb[i]);
  }
#endif

  // ---- sparse decode: each thread owns 4 output dims ----
  const int ns = (totalSel < KTOP) ? totalSel : KTOP;
  const int j0 = t * 4;
  float4 a        = *(const float4*)(b_dec + j0);
  const float4 bp = *(const float4*)(b_pre + j0);
  a.x += bp.x; a.y += bp.y; a.z += bp.z; a.w += bp.w;
  for (int s = 0; s < ns; ++s) {
    if (s + 4 < ns)
      __builtin_prefetch(W_dec + (size_t)selI[s + 4] * D_ + j0, 0, 0);
    const float  v  = selV[s];
    const float4 wv = *(const float4*)(W_dec + (size_t)selI[s] * D_ + j0);
    a.x = fmaf(v, wv.x, a.x);
    a.y = fmaf(v, wv.y, a.y);
    a.z = fmaf(v, wv.z, a.z);
    a.w = fmaf(v, wv.w, a.w);
  }
  *(float4*)(recon + (size_t)row * D_ + j0) = a;
}

// ---------------------------------------------------------------------------
extern "C" void kernel_launch(void* const* d_in, const int* in_sizes, int n_in,
                              void* d_out, int out_size, void* d_ws, size_t ws_size,
                              hipStream_t stream) {
  (void)in_sizes; (void)n_in; (void)out_size; (void)d_ws; (void)ws_size;
  const float* x     = (const float*)d_in[0];
  const float* b_pre = (const float*)d_in[1];
  const float* W_enc = (const float*)d_in[2];
  const float* b_enc = (const float*)d_in[3];
  const float* W_dec = (const float*)d_in[4];
  const float* b_dec = (const float*)d_in[5];

  float* recon   = (float*)d_out;                       // [B, D]
  float* latents = (float*)d_out + (size_t)B_ * D_;     // [B, L] (sparse_latents)

  dim3 g1(L_ / TN, B_ / TM);   // (64, 256)
  sae_encode_wmma<<<g1, 256, 0, stream>>>(x, b_pre, W_enc, b_enc, latents);
  sae_topk_decode<<<B_, 256, 0, stream>>>(b_pre, W_dec, b_dec, recon, latents);
}
